// FullyConnectedTensorProduct_59992103190608
// MI455X (gfx1250) — matrix operand order repaired and checked by reading, not compiled, not run on hardware
//
#include <hip/hip_runtime.h>

// ---------------------------------------------------------------------------
// FullyConnectedTensorProduct (e3nn-style), MUL=8, B=2,000,000, DIM=32.
//
// out[b, 0:32] = F[b, 0:512] @ Wbig[512, 32]
//   F features  : [s1 u x s2 v | dot(v1 u, v2 v) | s1 u * v2 v,k | v1 u,k * s2 v]
//   Wbig        : block-sparse repack of w_ss / (1/sqrt3) w_vv / w_sv / w_vs
//                 (v_out[w,k] -> output column 8 + 3w + k)
//
// GEMM tile: 16 batch rows x 32 output cols, K=512 via 16 steps of
// v_wmma_f32_16x16x32_f16 on two N-tiles (cols 0-15, 16-31), with a 1-step
// software pipeline: A(j+1) features are built before WMMA(j) issues, so the
// WMMA->VALU WAR hazard slots are filled with useful work instead of v_nops,
// and B(j) ds_loads are covered by feature math.
//
// A-fragment (16-bit A 16x32, ISA 7.12.2): lane L<16 -> row L, K slots
//   {0..7,16..23}; lane L+16 -> row L, K slots {8..15,24..31}. Both half-waves
//   hold the same row. The +8 K shift for hi lanes equals a u -> u+1 channel
//   shift (u0 always even), applied AT LOAD TIME with lane-dependent
//   addresses (no register-array selects -> no cndmask chains).
// Features are built natively in packed f16: each A VGPR holds features
//   (f, f+1) with f even -> one v_pk_mul_f16 / 3x v_pk_fma_f16 per VGPR.
// B-fragment (same K slots as A): lane L<16 -> col L, K {0..7,16..23}; lane L+16 ->
//   col L-16, K {8..15,24..31}. Prebuilt in fragment order by build_wfrag into d_ws.
// ---------------------------------------------------------------------------

typedef __attribute__((ext_vector_type(16))) _Float16 v16h;
typedef __attribute__((ext_vector_type(2)))  _Float16 h2;
typedef __attribute__((ext_vector_type(8)))  float    v8f;

#define TP_BATCH   2000000
#define TP_TILE_M  16
#define TP_NTILES  (TP_BATCH / TP_TILE_M)     // 125000, exact
#define TP_INV3    0.57735026918962576451f    // 1/sqrt(3)
#define TP_WFRAG_HALFS 16384                  // 16 j * 2 t * 32 lanes * 16 e

// ---------------------------------------------------------------------------
// Prep: build Wbig[512][32] (f16) directly in B-fragment order:
//   wfrag[j][t][lane][e] = Wbig[32*j + ((e<8 ? e : e+8) + 8*(lane>=16))][16*t + (lane&15)]
// ---------------------------------------------------------------------------
__global__ void build_wfrag(const float* __restrict__ w_ss,
                            const float* __restrict__ w_vv,
                            const float* __restrict__ w_sv,
                            const float* __restrict__ w_vs,
                            _Float16* __restrict__ wfrag) {
  int idx = blockIdx.x * blockDim.x + threadIdx.x;   // 0..16383
  if (idx >= TP_WFRAG_HALFS) return;
  int e = idx & 15;
  int L = (idx >> 4) & 31;
  int t = (idx >> 9) & 1;
  int j = idx >> 10;

  int K = (e < 8 ? e : e + 8) + ((L >= 16) ? 8 : 0);   // B-fragment K position
  int f = 32 * j + K;                 // Wbig row (feature index)
  int c = 16 * t + (L & 15);          // Wbig col (output channel)

  float val = 0.0f;
  if (f < 64) {                                        // ss
    int u = f >> 3, v = f & 7;
    if (c < 8) val = w_ss[(u * 8 + v) * 8 + c];
  } else if (f < 128) {                                // vv
    int g = f - 64;
    int u = g >> 3, v = g & 7;
    if (c < 8) val = TP_INV3 * w_vv[(u * 8 + v) * 8 + c];
  } else if (f < 320) {                                // sv -> col 8+3w+k
    int g = f - 128;
    int k = g >> 6, u = (g >> 3) & 7, v = g & 7;
    if (c >= 8 && ((c - 8) % 3) == k) val = w_sv[(u * 8 + v) * 8 + ((c - 8) / 3)];
  } else {                                             // vs -> col 8+3w+k
    int g = f - 320;
    int k = g >> 6, u = (g >> 3) & 7, v = g & 7;
    if (c >= 8 && ((c - 8) % 3) == k) val = w_vs[(u * 8 + v) * 8 + ((c - 8) / 3)];
  }
  wfrag[idx] = (_Float16)val;
}

// ---------------------------------------------------------------------------
// Main kernel: persistent blocks, 8 waves/block, one 16-row tile per wave
// per grid-stride iteration.
// ---------------------------------------------------------------------------
__global__ void __launch_bounds__(256)
ftp_wmma_kernel(const float* __restrict__ x1,
                const float* __restrict__ x2,
                const _Float16* __restrict__ wfrag,
                float* __restrict__ out) {
  __shared__ __align__(32) _Float16 ldsW[TP_WFRAG_HALFS];   // 32 KB

  // Stage all B fragments into LDS once per block.
  {
    const uint4* src = reinterpret_cast<const uint4*>(wfrag);
    uint4* dst = reinterpret_cast<uint4*>(ldsW);
    for (int i = threadIdx.x; i < TP_WFRAG_HALFS / 8; i += 256) dst[i] = src[i];
  }
  __syncthreads();

  const int lane = threadIdx.x & 31;
  const int wave = threadIdx.x >> 5;
  const int l    = lane & 15;      // row within tile / output col within N-tile
  const int hi   = lane >> 4;      // half-wave selector (A K-slots +8)

  const int waveGlobal = blockIdx.x * 8 + wave;
  const int waveStride = gridDim.x * 8;

  for (int tile = waveGlobal; tile < TP_NTILES; tile += waveStride) {
    const int row = tile * TP_TILE_M + l;
    const float* x1r = x1 + (size_t)row * 32;
    const float* x2r = x2 + (size_t)row * 32;

    // Prefetch next grid-stride tile's rows (global_prefetch_b8).
    if (tile + waveStride < TP_NTILES) {
      const size_t nrow = (size_t)((tile + waveStride) * TP_TILE_M + l) * 32;
      __builtin_prefetch(x1 + nrow, 0, 1);
      __builtin_prefetch(x2 + nrow, 0, 1);
    }

    // x1: load only this half-wave's u-parity directly (hi-shifted addresses).
    float suf[4], vuf[4][3];
#pragma unroll
    for (int c = 0; c < 4; ++c) {
      suf[c] = x1r[2 * c + hi];
      const float* pv = x1r + 8 + 6 * c + 3 * hi;
      vuf[c][0] = pv[0];
      vuf[c][1] = pv[1];
      vuf[c][2] = pv[2];
    }

    // x2: full row (8x b128); all indices below are compile-time constants.
    float r2[32];
    {
      const float4* p2 = reinterpret_cast<const float4*>(x2r);
#pragma unroll
      for (int q = 0; q < 8; ++q) {
        float4 b = p2[q];
        r2[4 * q + 0] = b.x; r2[4 * q + 1] = b.y;
        r2[4 * q + 2] = b.z; r2[4 * q + 3] = b.w;
      }
    }

    // One-time f16 conversion of the tile inputs.
    h2 hs2p[4], hv2t[3][4], hsu[4], hvu[4][3];
#pragma unroll
    for (int c = 0; c < 4; ++c) {
      hs2p[c].x = (_Float16)r2[2 * c];
      hs2p[c].y = (_Float16)r2[2 * c + 1];
      hsu[c].x = hsu[c].y = (_Float16)suf[c];
#pragma unroll
      for (int i = 0; i < 3; ++i) {
        hv2t[i][c].x = (_Float16)r2[8 + 6 * c + i];
        hv2t[i][c].y = (_Float16)r2[8 + 6 * c + 3 + i];
        hvu[c][i].x = hvu[c][i].y = (_Float16)vuf[c][i];
      }
    }

    union AF { v16h v; h2 h[8]; };

    // A fragment for K-step j: VGPR p holds feature pair (f, f+1),
    //   f = 32j + (p<4 ? 2p : 2p+8) + 8*hi  (hi shift folded into hsu/hvu).
    auto buildA = [&](int j, AF& a) {
#pragma unroll
      for (int p = 0; p < 8; ++p) {
        const int hb = p >> 2;          // K-block within chunk (u0 +2)
        const int c2 = p & 3;           // v-pair index
        h2 pr;
        if (j < 2) {                                   // ss
          pr = hsu[2 * j + hb] * hs2p[c2];
        } else if (j < 4) {                            // vv (dot over i)
          const int uc = 2 * (j - 2) + hb;
          pr = hvu[uc][0] * hv2t[0][c2];
          pr += hvu[uc][1] * hv2t[1][c2];
          pr += hvu[uc][2] * hv2t[2][c2];
        } else if (j < 10) {                           // sv
          const int jj = j - 4;
          pr = hsu[(2 * jj + hb) & 3] * hv2t[jj >> 1][c2];
        } else {                                       // vs
          const int jj = j - 10;
          pr = hvu[(2 * jj + hb) & 3][jj >> 1] * hs2p[c2];
        }
        a.h[p] = pr;
      }
    };

    v8f acc0 = {};   // output cols 0..15
    v8f acc1 = {};   // output cols 16..31

    AF aCur;
    buildA(0, aCur);

#pragma unroll
    for (int j = 0; j < 16; ++j) {
      // B fragments for this step (32B per lane, aligned).
      const v16h b0 = *reinterpret_cast<const v16h*>(ldsW + (j * 2 + 0) * 512 + lane * 16);
      const v16h b1 = *reinterpret_cast<const v16h*>(ldsW + (j * 2 + 1) * 512 + lane * 16);

      // Build NEXT step's A before issuing this step's WMMAs: fills the
      // WMMA->VALU WAR hazard window with useful pk-f16 work (no v_nops)
      // and covers the ds_load latency above.
      AF aNext;
      if (j < 15) buildA(j + 1, aNext);

      acc0 = __builtin_amdgcn_wmma_f32_16x16x32_f16(false, aCur.v, false, b0,
                                                    (short)0, acc0, false, false);
      acc1 = __builtin_amdgcn_wmma_f32_16x16x32_f16(false, aCur.v, false, b1,
                                                    (short)0, acc1, false, false);
      if (j < 15) aCur = aNext;
#if __has_builtin(__builtin_amdgcn_sched_barrier)
      // Pin VALU/WMMA per K-step (unconstrained reordering blew past 256
      // VGPRs in round 1), but let VMEM reads (0x10) and DS reads (0x40)
      // hoist across step boundaries.
      __builtin_amdgcn_sched_barrier(0x50);
#endif
    }

    // D layout: VGPR r -> row r (lanes 0-15) / row 8+r (lanes 16-31), col = l.
    float* orow = out + (size_t)(tile * TP_TILE_M + hi * 8) * 32;
#pragma unroll
    for (int r = 0; r < 8; ++r) {
      orow[(size_t)r * 32 + l]      = acc0[r];
      orow[(size_t)r * 32 + 16 + l] = acc1[r];
    }
  }
}

// ---------------------------------------------------------------------------
extern "C" void kernel_launch(void* const* d_in, const int* in_sizes, int n_in,
                              void* d_out, int out_size, void* d_ws, size_t ws_size,
                              hipStream_t stream) {
  const float* x1   = (const float*)d_in[0];
  const float* x2   = (const float*)d_in[1];
  const float* w_ss = (const float*)d_in[2];
  const float* w_vv = (const float*)d_in[3];
  const float* w_sv = (const float*)d_in[4];
  const float* w_vs = (const float*)d_in[5];

  _Float16* wfrag = (_Float16*)d_ws;          // needs 32 KB of d_ws
  float*    out   = (float*)d_out;

  // 1) Repack weights into WMMA B-fragment order (f16), 32 KB.
  build_wfrag<<<TP_WFRAG_HALFS / 256, 256, 0, stream>>>(w_ss, w_vv, w_sv, w_vs, wfrag);

  // 2) Persistent WMMA GEMM over 125000 16-row tiles.
  ftp_wmma_kernel<<<2048, 256, 0, stream>>>(x1, x2, wfrag, out);
}
